// RNN_4054449127908
// MI455X (gfx1250) — compile-verified, matrix operand order checked by reading, not run on hardware
//
#include <hip/hip_runtime.h>
#include <hip/hip_bf16.h>

typedef __attribute__((ext_vector_type(16))) __bf16 v16bf;
typedef __attribute__((ext_vector_type(8)))  __bf16 v8bf;
typedef __attribute__((ext_vector_type(8)))  float  v8f;

#define B_ 64
#define T_ 2048
#define I_ 64
#define H_ 128
#define K_ 8
#define TH_ (T_ * H_)                 // 262144
#define STATES_ELEMS (B_ * T_ * H_)   // 16777216
#define WFRAG_BYTES  (K_ * H_ * H_ * 2)   // 262144 bytes of bf16 W_hh fragments
#define SCAN_LDS_BYTES (WFRAG_BYTES + B_*H_*2 + B_*K_*4)  // 262144 + 16384 + 2048 = 280576

// -----------------------------------------------------------------------------
// Kernel 1: pre-swizzle W_hh (f32 [K,H,H]) into WMMA B-fragment order, bf16.
// Fragment f = ((kc*8 + gt)*4 + ks) holds the 32(K=h) x 16(N=g) bf16 tile for
// cell kc, output-column tile gt, k-step ks.  Stored as [frag][lane][elem] so
// the scan kernel loads each lane's v16bf with one contiguous 32B LDS read.
// Element mapping follows the ISA 16-bit operand layout: lanes 0-15 hold
// K 0-7 / 16-23 (elems 0-7 / 8-15), lanes 16-31 hold K 8-15 / 24-31.
// -----------------------------------------------------------------------------
__global__ void k_arrange_whh(const float* __restrict__ W_hh,
                              __bf16* __restrict__ wfrag) {
  int idx = blockIdx.x * blockDim.x + threadIdx.x;   // 0 .. 131071
  int e  = idx & 15;
  int l  = (idx >> 4) & 31;
  int f  = idx >> 9;
  int ks = f & 3;
  int gt = (f >> 2) & 7;
  int kc = f >> 5;
  int hi = l >> 4;
  int n  = l & 15;
  int Kl = (e & 7) + 8 * hi + ((e >= 8) ? 16 : 0);
  int g  = gt * 16 + n;        // output column (N)
  int h  = ks * 32 + Kl;       // contraction index (K = hidden dim)
  float v = W_hh[(kc * H_ + g) * H_ + h];   // B[h][g] = W_hh[kc][g][h]
  wfrag[f * 512 + l * 16 + e] = (__bf16)v;
}

// -----------------------------------------------------------------------------
// Kernel 2: fused effective-input projection.
//   W_ih_eff[b] = sum_k p[b,k] * W_ih[k]   (built in LDS, bank-pad stride 65)
//   u[b,t,g]    = x[b,t,:] . W_ih_eff[b,g,:] + b_eff[b,g]
// u is written straight into d_out's states slot [b,t,g]; the scan reads it
// there and overwrites it with the hidden state.
// grid = 64 batches x 8 time-chunks, 256 threads.
// -----------------------------------------------------------------------------
__global__ void k_uproj(const float* __restrict__ x,
                        const float* __restrict__ probs,
                        const float* __restrict__ W_ih,
                        const float* __restrict__ bias,
                        float* __restrict__ out) {
  __shared__ float Wl[H_ * 65];   // padded stride: stride-64 reads hit 1 bank
  __shared__ float bv[H_];
  int b  = blockIdx.x >> 3;
  int tc = blockIdx.x & 7;
  int tid = threadIdx.x;
  float p[K_];
#pragma unroll
  for (int k = 0; k < K_; ++k) p[k] = probs[b * K_ + k];
  for (int e = tid; e < H_ * I_; e += 256) {
    int g = e >> 6, i = e & 63;
    float s = 0.f;
#pragma unroll
    for (int k = 0; k < K_; ++k) s += p[k] * W_ih[(k * H_ + g) * I_ + i];
    Wl[g * 65 + i] = s;
  }
  if (tid < H_) {
    float s = 0.f;
#pragma unroll
    for (int k = 0; k < K_; ++k) s += p[k] * bias[k * H_ + tid];
    bv[tid] = s;
  }
  __syncthreads();
  int g    = tid & 127;
  int trow = tid >> 7;            // whole wave shares one t -> x loads scalarize
  int t0   = tc * 256;
  for (int tt = 0; tt < 256; tt += 2) {
    int t = t0 + tt + trow;
    const float* xr = &x[(b * T_ + t) * I_];
    float acc = bv[g];
#pragma unroll
    for (int i = 0; i < I_; ++i) acc += xr[i] * Wl[g * 65 + i];
    out[b * TH_ + t * H_ + g] = acc;
  }
}

// -----------------------------------------------------------------------------
// Kernel 3: the scan.  ONE workgroup (16 waves, 512 threads) on one WGP.
// All of W_hh lives in LDS as bf16 (256 KB of the 320 KB WGP LDS); the hidden
// state lives in LDS as bf16 (WMMA A operand).  Per step each wave computes
// two 16x16 output tiles of cand = h x W_hh^T for its g-column, chaining
// 4 k-steps per cell with v_wmma_f32_16x16x32_bf16, reusing each B fragment
// for both M tiles, then applies the per-row mixture weight p[b,k] and adds
// the precomputed fp32 u (read from d_out, then overwritten with h).
// 2048 steps x 1024 wmma = 2.1M wmma total, zero global sync in the scan.
// -----------------------------------------------------------------------------
__launch_bounds__(512, 1)
__global__ void k_scan(const float* __restrict__ h0,
                       const float* __restrict__ probs,
                       const __bf16* __restrict__ wfrag,
                       float* out) {
  extern __shared__ char smem[];
  __bf16* Wl  = (__bf16*)smem;                            // 256 KB fragments
  __bf16* hbf = (__bf16*)(smem + WFRAG_BYTES);            // 16 KB hidden state
  float*  pl  = (float*)(smem + WFRAG_BYTES + B_ * H_ * 2); // 2 KB probs
  int tid = threadIdx.x;

  {  // stage W fragments global -> LDS (once; 512 B per thread)
    const uint4* src = (const uint4*)wfrag;
    uint4* dst = (uint4*)Wl;
    for (int i = tid; i < WFRAG_BYTES / 16; i += 512) dst[i] = src[i];
  }
  for (int i = tid; i < B_ * H_; i += 512) hbf[i] = (__bf16)h0[i];
  for (int i = tid; i < B_ * K_; i += 512) pl[i] = probs[i];
  __syncthreads();

  int w  = tid >> 5;           // wave 0..15
  int l  = tid & 31;
  int gt = w & 7;              // g-column tile
  int mp = w >> 3;             // which pair of M (batch) tiles
  int m0 = mp * 2, m1 = mp * 2 + 1;
  int hi = l >> 4;
  int n  = l & 15;
  int g  = gt * 16 + n;
  int b0 = m0 * 16 + 8 * hi;   // acc element r maps to batch row b0 + r
  int b1 = m1 * 16 + 8 * hi;

  for (int t = 0; t < T_; ++t) {
    // fp32 u for this step (staged by k_uproj at this exact location)
    float acc0[8], acc1[8];
#pragma unroll
    for (int r = 0; r < 8; ++r) {
      acc0[r] = out[(b0 + r) * TH_ + t * H_ + g];
      acc1[r] = out[(b1 + r) * TH_ + t * H_ + g];
    }
    // build A fragments (h in bf16, ISA 16-bit A layout: row = lane&15,
    // K split 0-7/16-23 vs 8-15/24-31 by lane half -> two 16B LDS reads each)
    v16bf A0[4], A1[4];
#pragma unroll
    for (int ks = 0; ks < 4; ++ks) {
      int base = ks * 32 + 8 * hi;
      const __bf16* hr0 = &hbf[(m0 * 16 + n) * H_];
      const __bf16* hr1 = &hbf[(m1 * 16 + n) * H_];
      v8bf a0lo = *(const v8bf*)(hr0 + base);
      v8bf a0hi = *(const v8bf*)(hr0 + base + 16);
      v8bf a1lo = *(const v8bf*)(hr1 + base);
      v8bf a1hi = *(const v8bf*)(hr1 + base + 16);
      A0[ks] = __builtin_shufflevector(a0lo, a0hi, 0,1,2,3,4,5,6,7,8,9,10,11,12,13,14,15);
      A1[ks] = __builtin_shufflevector(a1lo, a1hi, 0,1,2,3,4,5,6,7,8,9,10,11,12,13,14,15);
    }

#pragma unroll
    for (int kc = 0; kc < K_; ++kc) {
      v8f d0 = {};
      v8f d1 = {};
#pragma unroll
      for (int ks = 0; ks < 4; ++ks) {
        const v16bf Bf =
            *(const v16bf*)&Wl[(((kc * 8 + gt) * 4 + ks) << 9) + l * 16];
        d0 = __builtin_amdgcn_wmma_f32_16x16x32_bf16(
            false, A0[ks], false, Bf, (short)0, d0, false, false);
        d1 = __builtin_amdgcn_wmma_f32_16x16x32_bf16(
            false, A1[ks], false, Bf, (short)0, d1, false, false);
      }
      // mixture weight p[b,kc] per output row, fp32 accumulate
#pragma unroll
      for (int r = 0; r < 8; ++r) {
        acc0[r] += pl[(b0 + r) * K_ + kc] * d0[r];
        acc1[r] += pl[(b1 + r) * K_ + kc] * d1[r];
      }
    }

    __syncthreads();   // every wave finished reading h (A fragments built)
#pragma unroll
    for (int r = 0; r < 8; ++r) {
      out[(b0 + r) * TH_ + t * H_ + g] = acc0[r];      // states[b,t,g]
      out[(b1 + r) * TH_ + t * H_ + g] = acc1[r];
      hbf[(b0 + r) * H_ + g] = (__bf16)acc0[r];        // next-step A operand
      hbf[(b1 + r) * H_ + g] = (__bf16)acc1[r];
    }
    if (t == T_ - 1) {
#pragma unroll
      for (int r = 0; r < 8; ++r) {
        out[STATES_ELEMS + (b0 + r) * H_ + g] = acc0[r];  // final hidden
        out[STATES_ELEMS + (b1 + r) * H_ + g] = acc1[r];
      }
    }
    __syncthreads();   // new h visible before next step's A build
  }
}

// -----------------------------------------------------------------------------
// Host launcher.  Input order: input, h_0, selection_probs, W_ih, W_hh, b.
// d_ws needs only 256 KB (W_hh bf16 fragments).
// -----------------------------------------------------------------------------
extern "C" void kernel_launch(void* const* d_in, const int* in_sizes, int n_in,
                              void* d_out, int out_size, void* d_ws, size_t ws_size,
                              hipStream_t stream) {
  const float* input = (const float*)d_in[0];
  const float* h_0   = (const float*)d_in[1];
  const float* probs = (const float*)d_in[2];
  const float* W_ih  = (const float*)d_in[3];
  const float* W_hh  = (const float*)d_in[4];
  const float* bias  = (const float*)d_in[5];
  float* out = (float*)d_out;
  __bf16* wfrag = (__bf16*)d_ws;

  (void)hipFuncSetAttribute((const void*)k_scan,
                            hipFuncAttributeMaxDynamicSharedMemorySize,
                            SCAN_LDS_BYTES);

  // 1) swizzle W_hh into WMMA B-fragment order (131072 elements)
  k_arrange_whh<<<512, 256, 0, stream>>>(W_hh, wfrag);
  // 2) effective input projection -> u staged in d_out states slots
  k_uproj<<<512, 256, 0, stream>>>(input, probs, W_ih, bias, out);
  // 3) single-workgroup LDS-resident WMMA scan
  k_scan<<<1, 512, SCAN_LDS_BYTES, stream>>>(h_0, probs, wfrag, out);
}
